// MessagePaiNN_53575422050759
// MI455X (gfx1250) — compile-verified, matrix operand order checked by reading, not run on hardware
//
#include <hip/hip_runtime.h>
#include <math.h>

typedef __attribute__((ext_vector_type(16))) _Float16 v16h;
typedef __attribute__((ext_vector_type(8)))  _Float16 v8h;
typedef __attribute__((ext_vector_type(8)))  float    v8f;

#define FEAT   128
#define COLS2  384
#define NRBF   20
#define CUTOFF 5.0f
#define PI_F   3.14159265358979323846f

// ---------------------------------------------------------------------------
// Kernel 0: seed d_out with node_scalar | node_vector (edge kernel accumulates
// deltas on top with atomics).
// ---------------------------------------------------------------------------
__global__ void painn_init_out(const float* __restrict__ node_scalar,
                               const float* __restrict__ node_vector,
                               float* __restrict__ out,
                               long long n_s, long long n_total) {
    long long i = (long long)blockIdx.x * blockDim.x + threadIdx.x;
    if (i >= n_total) return;
    out[i] = (i < n_s) ? node_scalar[i] : node_vector[i - n_s];
}

// ---------------------------------------------------------------------------
// Kernel 1: H = silu(X @ W1 + b1), f32 in, f16 out.  One wave = one 16x16 tile.
// ---------------------------------------------------------------------------
__global__ void painn_gemm1_silu(const float* __restrict__ X,
                                 const float* __restrict__ W1,
                                 const float* __restrict__ b1,
                                 _Float16* __restrict__ H,
                                 int nrows) {
    const int wave = (int)((blockIdx.x * blockDim.x + threadIdx.x) >> 5);
    const int lane = threadIdx.x & 31;
    const int colTiles = FEAT / 16;                 // 8
    const int rowTiles = (nrows + 15) >> 4;
    if (wave >= rowTiles * colTiles) return;

    const int tr = wave / colTiles;
    const int tc = wave % colTiles;
    const int laneLo = lane & 15;
    const int hi8 = (lane >> 4) << 3;               // lanes 16..31 -> +8 in K / +8 in M(out)
    int rowA = tr * 16 + laneLo;
    if (rowA >= nrows) rowA = nrows - 1;            // clamp; stores predicated below
    const int colB = tc * 16 + laneLo;

    v8f acc = {};
#pragma unroll
    for (int k0 = 0; k0 < FEAT; k0 += 32) {
        const float* xr = X + (size_t)rowA * FEAT + k0 + hi8;
        v16h a, b;
#pragma unroll
        for (int e = 0; e < 16; ++e) {
            const int kk = ((e >> 3) << 4) + (e & 7);        // {0..7,16..23}
            a[e] = (_Float16)xr[kk];
            b[e] = (_Float16)W1[(size_t)(k0 + kk + hi8) * FEAT + colB];
        }
        acc = __builtin_amdgcn_wmma_f32_16x16x32_f16(false, a, false, b,
                                                     (short)0, acc, false, false);
    }

    const float bias = b1[colB];
#pragma unroll
    for (int r = 0; r < 8; ++r) {
        const int row = tr * 16 + r + hi8;
        if (row < nrows) {
            const float c = acc[r] + bias;
            const float s = c / (1.0f + __expf(-c));         // silu
            H[(size_t)row * FEAT + colB] = (_Float16)s;
        }
    }
}

// ---------------------------------------------------------------------------
// Kernel 2: atom_scalar = H @ W2 + b2, f16 in, f32 out (cols = 384).
// ---------------------------------------------------------------------------
__global__ void painn_gemm2(const _Float16* __restrict__ H,
                            const float* __restrict__ W2,
                            const float* __restrict__ b2,
                            float* __restrict__ A,
                            int nrows) {
    const int wave = (int)((blockIdx.x * blockDim.x + threadIdx.x) >> 5);
    const int lane = threadIdx.x & 31;
    const int colTiles = COLS2 / 16;                // 24
    const int rowTiles = (nrows + 15) >> 4;
    if (wave >= rowTiles * colTiles) return;

    const int tr = wave / colTiles;
    const int tc = wave % colTiles;
    const int laneLo = lane & 15;
    const int hi8 = (lane >> 4) << 3;
    int rowA = tr * 16 + laneLo;
    if (rowA >= nrows) rowA = nrows - 1;
    const int colB = tc * 16 + laneLo;

    v8f acc = {};
#pragma unroll
    for (int k0 = 0; k0 < FEAT; k0 += 32) {
        const _Float16* hr = H + (size_t)rowA * FEAT + k0 + hi8;
        const v8h p0 = *reinterpret_cast<const v8h*>(hr);        // K+0..7
        const v8h p1 = *reinterpret_cast<const v8h*>(hr + 16);   // K+16..23
        v16h a, b;
#pragma unroll
        for (int e = 0; e < 8; ++e) { a[e] = p0[e]; a[e + 8] = p1[e]; }
#pragma unroll
        for (int e = 0; e < 16; ++e) {
            const int kk = ((e >> 3) << 4) + (e & 7);
            b[e] = (_Float16)W2[(size_t)(k0 + kk + hi8) * COLS2 + colB];
        }
        acc = __builtin_amdgcn_wmma_f32_16x16x32_f16(false, a, false, b,
                                                     (short)0, acc, false, false);
    }

    const float bias = b2[colB];
#pragma unroll
    for (int r = 0; r < 8; ++r) {
        const int row = tr * 16 + r + hi8;
        if (row < nrows)
            A[(size_t)row * COLS2 + colB] = acc[r] + bias;
    }
}

// ---------------------------------------------------------------------------
// Kernel 3: fused edge pass, RBF contraction on the matrix pipe.
// Block = 256 threads = 8 waves; one iteration handles 16 edges.
//   A (16x32 f16) = sinc basis, rows=edges, K=basis (padded 20->32 w/ zeros)
//   B (32x16 f16) = Wr column tiles; wave w owns col tiles {16w,128+16w,256+16w}
//     so lane feature f = 16w + (lane&15) gets its s1/s2/s3 from its own accs.
// B fragments are preloaded once and reused across a grid-stride group loop.
// ---------------------------------------------------------------------------
__global__ void painn_edge_wmma(const float* __restrict__ adj,         // [E,6]
                                const float* __restrict__ atom,        // [N,384]
                                const float* __restrict__ node_vector, // [N,128,3]
                                const float* __restrict__ Wr,          // [20,384]
                                const float* __restrict__ br,          // [384]
                                float* __restrict__ out_s,             // [N,128]
                                float* __restrict__ out_v,             // [N,128,3]
                                int E, int nGroups) {
    const int lane   = threadIdx.x & 31;
    const int wv     = threadIdx.x >> 5;            // 0..7
    const int laneLo = lane & 15;
    const int hi8    = (lane >> 4) << 3;
    const int f      = wv * 16 + laneLo;            // feature in [0,128)

    // ---- preload Wr B-fragments (rows n>=20 zero-padded) -------------------
    v16h bw1, bw2, bw3;
#pragma unroll
    for (int e = 0; e < 16; ++e) {
        const int n = ((e >> 3) << 4) + (e & 7) + hi8;   // K row
        const bool v = (n < NRBF);
        bw1[e] = (_Float16)(v ? Wr[n * COLS2 + f]            : 0.0f);
        bw2[e] = (_Float16)(v ? Wr[n * COLS2 + FEAT + f]     : 0.0f);
        bw3[e] = (_Float16)(v ? Wr[n * COLS2 + 2 * FEAT + f] : 0.0f);
    }
    const float br1 = br[f], br2 = br[FEAT + f], br3 = br[2 * FEAT + f];

    for (int g = blockIdx.x; g < nGroups; g += gridDim.x) {
        // each lane owns edge slot (lane&15); lanes 16..31 duplicate
        const int myEdge = g * 16 + laneLo;
        const int ld = (myEdge < E) ? myEdge : (E - 1);
        const float* a6 = adj + (size_t)ld * 6;
        const int   dstI = (int)a6[0];
        const int   srcI = (int)a6[1];
        const float d    = a6[5];
        const float inv_d = 1.0f / d;
        const float rh0 = a6[2] * inv_d;
        const float rh1 = a6[3] * inv_d;
        const float rh2 = a6[4] * inv_d;
        const float cut = (d < CUTOFF)
                        ? 0.5f * (__cosf(PI_F * d * (1.0f / CUTOFF)) + 1.0f)
                        : 0.0f;

        // ---- A fragment: sinc(d * n * pi/5)/d in WMMA A layout -------------
        v16h a;
#pragma unroll
        for (int e = 0; e < 16; ++e) {
            const int n = ((e >> 3) << 4) + (e & 7) + hi8;
            float s = 0.0f;
            if (n < NRBF)
                s = __sinf(d * (float)(n + 1) * (PI_F / CUTOFF)) * inv_d;
            a[e] = (_Float16)s;
        }

        v8f acc1 = {}, acc2 = {}, acc3 = {};
        acc1 = __builtin_amdgcn_wmma_f32_16x16x32_f16(false, a, false, bw1,
                                                      (short)0, acc1, false, false);
        acc2 = __builtin_amdgcn_wmma_f32_16x16x32_f16(false, a, false, bw2,
                                                      (short)0, acc2, false, false);
        acc3 = __builtin_amdgcn_wmma_f32_16x16x32_f16(false, a, false, bw3,
                                                      (short)0, acc3, false, false);

        // ---- per edge slot: gather, modulate, scatter-add ------------------
#pragma unroll
        for (int r = 0; r < 8; ++r) {
            const int m  = r + hi8;                 // edge slot of this acc row
            const int em = g * 16 + m;
            const int   sSrc = __shfl(srcI, m, 32);
            const int   sDst = __shfl(dstI, m, 32);
            const float sCut = __shfl(cut,  m, 32);
            const float sRh0 = __shfl(rh0,  m, 32);
            const float sRh1 = __shfl(rh1,  m, 32);
            const float sRh2 = __shfl(rh2,  m, 32);
            if (em < E) {
                const float w1 = (acc1[r] + br1) * sCut;
                const float w2 = (acc2[r] + br2) * sCut;
                const float w3 = (acc3[r] + br3) * sCut;
                const float* arow = atom + (size_t)sSrc * COLS2;
                const float p1 = arow[f] * w1;
                const float p2 = arow[FEAT + f] * w2;
                const float p3 = arow[2 * FEAT + f] * w3;

                atomicAdd(out_s + (size_t)sDst * FEAT + f, p2);

                const float* nv = node_vector + (size_t)sSrc * COLS2 + f * 3;
                float* ov = out_v + (size_t)sDst * COLS2 + f * 3;
                atomicAdd(&ov[0], fmaf(nv[0], p1, p3 * sRh0));
                atomicAdd(&ov[1], fmaf(nv[1], p1, p3 * sRh1));
                atomicAdd(&ov[2], fmaf(nv[2], p1, p3 * sRh2));
            }
        }
    }
}

// ---------------------------------------------------------------------------
extern "C" void kernel_launch(void* const* d_in, const int* in_sizes, int n_in,
                              void* d_out, int out_size, void* d_ws, size_t ws_size,
                              hipStream_t stream) {
    const float* node_scalar = (const float*)d_in[0];   // [N,128]
    const float* node_vector = (const float*)d_in[1];   // [N,128,3]
    const float* adj         = (const float*)d_in[2];   // [E,6]
    const float* W1          = (const float*)d_in[3];   // [128,128]
    const float* b1          = (const float*)d_in[4];   // [128]
    const float* W2          = (const float*)d_in[5];   // [128,384]
    const float* b2          = (const float*)d_in[6];   // [384]
    const float* Wr          = (const float*)d_in[7];   // [20,384]
    const float* br          = (const float*)d_in[8];   // [384]

    const int N = in_sizes[0] / FEAT;
    const int E = in_sizes[2] / 6;

    // workspace layout: H (f16, N*128) | atom_scalar (f32, N*384)
    _Float16* Hws = (_Float16*)d_ws;
    size_t h_bytes = ((size_t)N * FEAT * sizeof(_Float16) + 255) & ~(size_t)255;
    float* atom = (float*)((char*)d_ws + h_bytes);

    float* out_s = (float*)d_out;                // [N,128]
    float* out_v = out_s + (size_t)N * FEAT;     // [N,128,3]

    // 0) seed output
    {
        const long long n_s = (long long)N * FEAT;
        const long long n_total = n_s * 4;       // + N*128*3
        const int thr = 256;
        const long long blk = (n_total + thr - 1) / thr;
        painn_init_out<<<(unsigned)blk, thr, 0, stream>>>(node_scalar, node_vector,
                                                          out_s, n_s, n_total);
    }
    // 1) H = silu(X @ W1 + b1)
    {
        const int rowTiles = (N + 15) >> 4;
        const int waves = rowTiles * (FEAT / 16);
        const int thr = 256;
        const int blk = (waves * 32 + thr - 1) / thr;
        painn_gemm1_silu<<<blk, thr, 0, stream>>>(node_scalar, W1, b1, Hws, N);
    }
    // 2) atom_scalar = H @ W2 + b2
    {
        const int rowTiles = (N + 15) >> 4;
        const int waves = rowTiles * (COLS2 / 16);
        const int thr = 256;
        const int blk = (waves * 32 + thr - 1) / thr;
        painn_gemm2<<<blk, thr, 0, stream>>>(Hws, W2, b2, atom, N);
    }
    // 3) fused edge pass: WMMA RBF + gather + scatter-add
    {
        const int nGroups = (E + 15) / 16;
        const int thr = 256;
        int blk = nGroups < 4096 ? nGroups : 4096;
        if (blk < 1) blk = 1;
        painn_edge_wmma<<<blk, thr, 0, stream>>>(adj, atom, node_vector, Wr, br,
                                                 out_s, out_v, E, nGroups);
    }
}
